// ChannelAttention_29729763623404
// MI455X (gfx1250) — matrix operand order repaired and checked by reading, not compiled, not run on hardware
//
#include <hip/hip_runtime.h>
#include <hip/hip_bf16.h>

// ---------------------------------------------------------------------------
// Channel attention: att = softmax_c(X X^T); out = beta * (att X) + x
// B=4, C=128, H=W=256 -> N=65536.  Matmuls via v_wmma_f32_16x16x32_bf16.
//
// Fast path (needs ~128.6 MB scratch): pre-pack x into bf16 twice
//   Xbf [B][C][N]  row-major   -> Gram A/B operands are contiguous b128 loads
//   XT  [B][N][C]  transposed  -> apply B operand is one contiguous v16bf load
// Fallback path (tiny scratch): round-1 inline fp32->bf16 convert kernels.
// ---------------------------------------------------------------------------

#define CHN   128
#define NPIX  65536          // 256*256
#define BATCH 4
#define KSPLIT 32            // K-slabs for the Gram reduction
#define KSLAB  (NPIX / KSPLIT)   // 2048
#define TPX   64             // pixels per pack-kernel block

typedef __attribute__((ext_vector_type(16))) __bf16 v16bf;
typedef __attribute__((ext_vector_type(8)))  __bf16 v8bf;
typedef __attribute__((ext_vector_type(8)))  float  v8f;

// ---------------------------------------------------------------------------
// Kernel 0: zero the Gram scratch (d_ws is poisoned 0xAA by the harness)
// ---------------------------------------------------------------------------
__global__ void zero_kernel(float* __restrict__ p, int n) {
    int i = blockIdx.x * blockDim.x + threadIdx.x;
    if (i < n) p[i] = 0.0f;
}

// ---------------------------------------------------------------------------
// Kernel P: pack x (fp32, [B][C][N]) into bf16 row-major Xbf [B][C][N] and
// bf16 transposed XT [B][N][C].  One 128(C) x 64(N) tile per block via LDS.
// ---------------------------------------------------------------------------
__global__ __launch_bounds__(256)
void pack_kernel(const float* __restrict__ x,
                 __bf16* __restrict__ xbf, __bf16* __restrict__ xt) {
    __shared__ float tile[CHN][TPX + 1];        // +1 pad: conflict-free transpose
    int b  = blockIdx.y;
    int n0 = blockIdx.x * TPX;
    int t  = threadIdx.x;
    const float* X = x + (size_t)b * CHN * NPIX;

#pragma unroll
    for (int i = 0; i < (CHN * TPX) / 256; ++i) {   // 32 iters
        int idx = i * 256 + t;
        int c = idx >> 6, j = idx & 63;
        tile[c][j] = X[(size_t)c * NPIX + n0 + j];
    }
    __syncthreads();

    __bf16* Xb = xbf + (size_t)b * CHN * NPIX;
#pragma unroll
    for (int i = 0; i < (CHN * TPX) / 256; ++i) {
        int idx = i * 256 + t;
        int c = idx >> 6, j = idx & 63;
        Xb[(size_t)c * NPIX + n0 + j] = (__bf16)tile[c][j];
    }
    __bf16* XT = xt + (size_t)b * NPIX * CHN;
#pragma unroll
    for (int i = 0; i < (TPX * CHN) / 256; ++i) {
        int idx = i * 256 + t;
        int j = idx >> 7, c = idx & 127;            // contiguous along c
        XT[(size_t)(n0 + j) * CHN + c] = (__bf16)tile[c][j];
    }
}

// ---------------------------------------------------------------------------
// WMMA operand layouts (CDNA5 ISA 7.12.2, wave32), h = lane/16:
//   A 16x32 bf16: lane row = lane%16; elems 0..15 -> K = h*16 + 0..15 (this code's order, not ISA table)
//   B 32x16 bf16: lane col = lane%16; elems 0..15 -> K = h*16 + 0..15 (this code's order, not ISA table)
//   C/D 16x16 f32: lane col = lane%16; VGPR p -> row m = p + h*8
// ---------------------------------------------------------------------------
__device__ __forceinline__ v16bf make_a(const __bf16* row, int k, int lhalf) {
    v8bf lo = *(const v8bf*)(row + k + lhalf * 16);
    v8bf hi = *(const v8bf*)(row + k + lhalf * 16 + 8);
    return __builtin_shufflevector(lo, hi, 0,1,2,3,4,5,6,7,8,9,10,11,12,13,14,15);
}

// ---------------------------------------------------------------------------
// Kernel 1 (fast): Gram matrix att_raw[b,c,d] = sum_n X[b,c,n]*X[b,d,n]
// One wave per (batch, 16x16 tile, K-slab); slabs combined with f32 atomics.
// Inner loop: 2+2 b128 loads + 1 WMMA, zero converts.
// ---------------------------------------------------------------------------
__global__ __launch_bounds__(32)
void gram_kernel_bf(const __bf16* __restrict__ xbf, float* __restrict__ att_raw) {
    int bid  = blockIdx.x;
    int kc   = bid % KSPLIT;
    int tile = (bid / KSPLIT) % 64;
    int b    = bid / (KSPLIT * 64);
    int ti = tile >> 3, tj = tile & 7;

    int lane = threadIdx.x, lmod = lane & 15, lhalf = lane >> 4;

    const __bf16* Xb   = xbf + (size_t)b * CHN * NPIX;
    const __bf16* Arow = Xb + (size_t)(ti * 16 + lmod) * NPIX;
    const __bf16* Brow = Xb + (size_t)(tj * 16 + lmod) * NPIX;   // B[k,n]=X[d,k]

    v8f acc = {};
    int k0 = kc * KSLAB;
    for (int k = k0; k < k0 + KSLAB; k += 32) {
        __builtin_prefetch((const void*)(Arow + k + 512), 0, 1);  // global_prefetch_b8
        __builtin_prefetch((const void*)(Brow + k + 512), 0, 1);
        v16bf av = make_a(Arow, k, lhalf);
        v16bf bv = *(const v16bf*)(Brow + k + lhalf * 16);
        acc = __builtin_amdgcn_wmma_f32_16x16x32_bf16(
                  false, av, false, bv, (short)0, acc, false, false);
    }
    float* T = att_raw + ((size_t)b * CHN + ti * 16) * CHN + tj * 16;
#pragma unroll
    for (int p = 0; p < 8; ++p)
        atomicAdd(&T[(size_t)(p + lhalf * 8) * CHN + lmod], acc[p]);
}

// ---------------------------------------------------------------------------
// Kernel 1 (fallback): inline fp32->bf16 convert variant (round-1 code)
// ---------------------------------------------------------------------------
__global__ __launch_bounds__(32)
void gram_kernel_f32(const float* __restrict__ x, float* __restrict__ att_raw) {
    int bid  = blockIdx.x;
    int kc   = bid % KSPLIT;
    int tile = (bid / KSPLIT) % 64;
    int b    = bid / (KSPLIT * 64);
    int ti = tile >> 3, tj = tile & 7;
    int lane = threadIdx.x, lmod = lane & 15, lhalf = lane >> 4;

    const float* X    = x + (size_t)b * CHN * NPIX;
    const float* Arow = X + (size_t)(ti * 16 + lmod) * NPIX;
    const float* Brow = X + (size_t)(tj * 16 + lmod) * NPIX;

    v8f acc = {};
    int k0 = kc * KSLAB;
    for (int k = k0; k < k0 + KSLAB; k += 32) {
        v16bf av, bv;
#pragma unroll
        for (int p = 0; p < 8; ++p) {
            int kk = lhalf * 16 + 2 * p;
            av[2 * p]     = (__bf16)Arow[k + kk];
            av[2 * p + 1] = (__bf16)Arow[k + kk + 1];
        }
#pragma unroll
        for (int p = 0; p < 8; ++p) {
            int kk = lhalf * 16 + 2 * p;
            bv[2 * p]     = (__bf16)Brow[k + kk];
            bv[2 * p + 1] = (__bf16)Brow[k + kk + 1];
        }
        acc = __builtin_amdgcn_wmma_f32_16x16x32_bf16(
                  false, av, false, bv, (short)0, acc, false, false);
    }
    float* T = att_raw + ((size_t)b * CHN + ti * 16) * CHN + tj * 16;
#pragma unroll
    for (int p = 0; p < 8; ++p)
        atomicAdd(&T[(size_t)(p + lhalf * 8) * CHN + lmod], acc[p]);
}

// ---------------------------------------------------------------------------
// Kernel 2: softmax over axis=1 (query-channel axis c) per (b,d) column.
// Emits both f32 (fallback) and bf16 (fast apply) copies.
// ---------------------------------------------------------------------------
__global__ __launch_bounds__(CHN)
void softmax_kernel(const float* __restrict__ att_raw,
                    float* __restrict__ att_s, __bf16* __restrict__ att_bf) {
    int b = blockIdx.x;
    int d = threadIdx.x;
    const float* col = att_raw + (size_t)b * CHN * CHN + d;

    float m = -INFINITY;
#pragma unroll 4
    for (int c = 0; c < CHN; ++c) m = fmaxf(m, col[c * CHN]);
    float s = 0.0f;
#pragma unroll 4
    for (int c = 0; c < CHN; ++c) s += __expf(col[c * CHN] - m);
    float inv = 1.0f / s;

    float*  of = att_s  + (size_t)b * CHN * CHN + d;
    __bf16* ob = att_bf + (size_t)b * CHN * CHN + d;
#pragma unroll 4
    for (int c = 0; c < CHN; ++c) {
        float v = __expf(col[c * CHN] - m) * inv;
        of[c * CHN] = v;
        ob[c * CHN] = (__bf16)v;
    }
}

// ---------------------------------------------------------------------------
// Kernel 3 (fast): xw = att * X via bf16 operands; out = beta*xw + x.
// One wave per 16(M) x 64(N) tile; A reused across 4 N-subtiles; B operand is
// one contiguous v16bf load from the transposed XT copy.
// ---------------------------------------------------------------------------
__global__ __launch_bounds__(32)
void apply_kernel_bf(const float* __restrict__ x, const __bf16* __restrict__ att_bf,
                     const __bf16* __restrict__ xt, const float* __restrict__ beta_p,
                     float* __restrict__ out) {
    int bid   = blockIdx.x;
    int ntile = bid % (NPIX / 64);
    int mtile = (bid / (NPIX / 64)) % (CHN / 16);
    int b     = bid / ((NPIX / 64) * (CHN / 16));
    int lane = threadIdx.x, lmod = lane & 15, lhalf = lane >> 4;

    float beta = beta_p[0];
    const __bf16* XT   = xt + (size_t)b * NPIX * CHN;
    const __bf16* Arow = att_bf + (size_t)b * CHN * CHN + (size_t)(mtile * 16 + lmod) * CHN;
    int c0 = mtile * 16, n0 = ntile * 64;

    v8f acc[4] = {v8f{}, v8f{}, v8f{}, v8f{}};
#pragma unroll
    for (int k = 0; k < CHN; k += 32) {
        v16bf av = make_a(Arow, k, lhalf);
#pragma unroll
        for (int t = 0; t < 4; ++t) {
            // B[k..k+31, n]: contiguous along channel in XT[pix][c]
            v16bf bv = *(const v16bf*)(XT + (size_t)(n0 + t * 16 + lmod) * CHN + k + lhalf * 16);
            acc[t] = __builtin_amdgcn_wmma_f32_16x16x32_bf16(
                         false, av, false, bv, (short)0, acc[t], false, false);
        }
    }
#pragma unroll
    for (int t = 0; t < 4; ++t)
#pragma unroll
        for (int p = 0; p < 8; ++p) {
            int m = c0 + p + lhalf * 8;
            size_t idx = (size_t)b * CHN * NPIX + (size_t)m * NPIX + n0 + t * 16 + lmod;
            out[idx] = beta * acc[t][p] + x[idx];
        }
}

// ---------------------------------------------------------------------------
// Kernel 3 (fallback): inline-convert variant (round-1 code)
// ---------------------------------------------------------------------------
__global__ __launch_bounds__(32)
void apply_kernel_f32(const float* __restrict__ x, const float* __restrict__ att_s,
                      const float* __restrict__ beta_p, float* __restrict__ out) {
    int bid   = blockIdx.x;
    int ntile = bid % (NPIX / 64);
    int mtile = (bid / (NPIX / 64)) % (CHN / 16);
    int b     = bid / ((NPIX / 64) * (CHN / 16));
    int lane = threadIdx.x, lmod = lane & 15, lhalf = lane >> 4;

    float beta = beta_p[0];
    const float* X = x + (size_t)b * CHN * NPIX;
    const float* Arow = att_s + (size_t)b * CHN * CHN + (size_t)(mtile * 16 + lmod) * CHN;
    int c0 = mtile * 16, n0 = ntile * 64;

    v8f acc[4] = {v8f{}, v8f{}, v8f{}, v8f{}};
#pragma unroll
    for (int k = 0; k < CHN; k += 32) {
        v16bf av;
#pragma unroll
        for (int p = 0; p < 8; ++p) {
            int kk = lhalf * 16 + 2 * p;
            av[2 * p]     = (__bf16)Arow[k + kk];
            av[2 * p + 1] = (__bf16)Arow[k + kk + 1];
        }
#pragma unroll
        for (int t = 0; t < 4; ++t) {
            v16bf bv;
#pragma unroll
            for (int p = 0; p < 8; ++p) {
                int kk = lhalf * 16 + 2 * p;
                bv[2 * p]     = (__bf16)X[(size_t)(k + kk)     * NPIX + n0 + t * 16 + lmod];
                bv[2 * p + 1] = (__bf16)X[(size_t)(k + kk + 1) * NPIX + n0 + t * 16 + lmod];
            }
            acc[t] = __builtin_amdgcn_wmma_f32_16x16x32_bf16(
                         false, av, false, bv, (short)0, acc[t], false, false);
        }
    }
#pragma unroll
    for (int t = 0; t < 4; ++t)
#pragma unroll
        for (int p = 0; p < 8; ++p) {
            int m = c0 + p + lhalf * 8;
            size_t idx = (size_t)b * CHN * NPIX + (size_t)m * NPIX + n0 + t * 16 + lmod;
            out[idx] = beta * acc[t][p] + x[idx];
        }
}

// ---------------------------------------------------------------------------
extern "C" void kernel_launch(void* const* d_in, const int* in_sizes, int n_in,
                              void* d_out, int out_size, void* d_ws, size_t ws_size,
                              hipStream_t stream) {
    const float* x    = (const float*)d_in[0];
    const float* beta = (const float*)d_in[1];
    float*       out  = (float*)d_out;

    const size_t xt_bytes  = (size_t)BATCH * NPIX * CHN * sizeof(__bf16);   // 64 MB
    const size_t xbf_bytes = xt_bytes;                                      // 64 MB
    const size_t att_elems = (size_t)BATCH * CHN * CHN;                     // 65536
    const size_t full_bytes = xt_bytes + xbf_bytes
                            + att_elems * (2 * sizeof(float) + sizeof(__bf16));

    if (ws_size >= full_bytes) {
        // ---- fast path: packed bf16 operands ----
        char* p = (char*)d_ws;
        __bf16* xt      = (__bf16*)p;               p += xt_bytes;
        __bf16* xbf     = (__bf16*)p;               p += xbf_bytes;
        float*  att_raw = (float*)p;                p += att_elems * sizeof(float);
        float*  att_s   = (float*)p;                p += att_elems * sizeof(float);
        __bf16* att_bf  = (__bf16*)p;

        pack_kernel<<<dim3(NPIX / TPX, BATCH), 256, 0, stream>>>(x, xbf, xt);
        zero_kernel<<<((int)att_elems + 255) / 256, 256, 0, stream>>>(att_raw, (int)att_elems);
        gram_kernel_bf<<<BATCH * 64 * KSPLIT, 32, 0, stream>>>(xbf, att_raw);
        softmax_kernel<<<BATCH, CHN, 0, stream>>>(att_raw, att_s, att_bf);
        apply_kernel_bf<<<BATCH * (CHN / 16) * (NPIX / 64), 32, 0, stream>>>(
            x, att_bf, xt, beta, out);
    } else {
        // ---- fallback: inline-convert path, tiny scratch ----
        char* p = (char*)d_ws;
        float*  att_raw = (float*)p;                p += att_elems * sizeof(float);
        float*  att_s   = (float*)p;                p += att_elems * sizeof(float);
        __bf16* att_bf  = (__bf16*)p;

        zero_kernel<<<((int)att_elems + 255) / 256, 256, 0, stream>>>(att_raw, (int)att_elems);
        gram_kernel_f32<<<BATCH * 64 * KSPLIT, 32, 0, stream>>>(x, att_raw);
        softmax_kernel<<<BATCH, CHN, 0, stream>>>(att_raw, att_s, att_bf);
        apply_kernel_f32<<<BATCH * (CHN / 16) * (NPIX / 64), 32, 0, stream>>>(
            x, att_s, beta, out);
    }
    (void)in_sizes; (void)n_in; (void)out_size;
}
